// AdvancedCNNLSTM_46806553592358
// MI455X (gfx1250) — matrix operand order repaired and checked by reading, not compiled, not run on hardware
//
#include <hip/hip_runtime.h>

// ---------------------------------------------------------------------------
// CDNA5 (gfx1250) implementation of AdvancedCNNLSTM forward pass.
// All dense layers run on v_wmma_f32_16x16x32_f16 (f16 in, f32 accum).
// GEMM variants are templated so the k-loop is guard-free (EXEC all-ones),
// the k-loop is ping-pong unrolled x2 (no register copies), and all GEMM
// memory traffic goes through explicit address_space(1) pointers so every
// access lowers to global_load_b128 / global_store_b32 (never flat_*).
// ---------------------------------------------------------------------------

typedef __attribute__((ext_vector_type(16))) _Float16 v16h;
typedef __attribute__((ext_vector_type(8)))  float    v8f;
typedef __attribute__((ext_vector_type(4)))  float    f4_t;
typedef _Float16 h16_t;

// global-address-space pointer types (built via inttoptr -> guaranteed AS1)
typedef const f4_t  __attribute__((address_space(1)))* gcf4p;
typedef const float __attribute__((address_space(1)))* gcfp;
typedef float       __attribute__((address_space(1)))* gfp;

#define EPSBN 1e-3f

// Hint for the simple elementwise kernels
#define ASSUME_GLOBAL(p)                                                    \
  __builtin_assume(!__builtin_amdgcn_is_shared((void*)(p)) &&               \
                   !__builtin_amdgcn_is_private((void*)(p)))

// ---------------- weight pack: W (K x N row-major f32) -> BT (Np x Kp f16, zero pad)
__global__ void k_pack_w(const float* __restrict__ W, h16_t* __restrict__ BT,
                         int K, int N, int Kp, int Np) {
  ASSUME_GLOBAL(W); ASSUME_GLOBAL(BT);
  long idx = (long)blockIdx.x * 256 + threadIdx.x;
  long tot = (long)Np * Kp;
  if (idx >= tot) return;
  int n = (int)(idx / Kp), k = (int)(idx % Kp);
  float v = (n < N && k < K) ? W[(long)k * N + n] : 0.f;
  BT[idx] = (h16_t)v;
}

// ---------------- vector pack with pad value
__global__ void k_pack_vec(const float* __restrict__ src, float* __restrict__ dst,
                           int N, int Np, float padval) {
  ASSUME_GLOBAL(src); ASSUME_GLOBAL(dst);
  int i = blockIdx.x * 256 + threadIdx.x;
  if (i >= Np) return;
  dst[i] = (i < N) ? src[i] : padval;
}

// ---------------- f32 slice -> f16 with zero pad to width Wout, col offset co
__global__ void k_cvt_f16(const float* __restrict__ X, int ldx, int c0, int K,
                          h16_t* __restrict__ Y, int ldy, int co, int Wout, long Mr) {
  ASSUME_GLOBAL(X); ASSUME_GLOBAL(Y);
  long idx = (long)blockIdx.x * 256 + threadIdx.x;
  long tot = Mr * (long)Wout;
  if (idx >= tot) return;
  long m = idx / Wout;
  int  k = (int)(idx % Wout);
  float v = (k < K) ? X[m * (long)ldx + c0 + k] : 0.f;
  Y[m * (long)ldy + co + k] = (h16_t)v;
}

__global__ void k_zero32(float* p, long n) {
  ASSUME_GLOBAL(p);
  long i = (long)blockIdx.x * 256 + threadIdx.x;
  if (i < n) p[i] = 0.f;
}
__global__ void k_zero16(h16_t* p, long n) {
  ASSUME_GLOBAL(p);
  long i = (long)blockIdx.x * 256 + threadIdx.x;
  if (i < n) p[i] = (h16_t)0.f;
}

// ---------------- batchnorm (input already relu'd in gemm epilogue) -> f16
__global__ void k_bn_f16(const float* __restrict__ X, int ldx,
                         const float* __restrict__ g, const float* __restrict__ b,
                         const float* __restrict__ mn, const float* __restrict__ vr,
                         h16_t* __restrict__ Y, int ldy, long Mr, int N) {
  ASSUME_GLOBAL(X); ASSUME_GLOBAL(Y);
  ASSUME_GLOBAL(g); ASSUME_GLOBAL(b); ASSUME_GLOBAL(mn); ASSUME_GLOBAL(vr);
  long idx = (long)blockIdx.x * 256 + threadIdx.x;
  long tot = Mr * (long)N;
  if (idx >= tot) return;
  long m = idx / N;
  int  n = (int)(idx % N);
  float x = X[m * (long)ldx + n];
  float y = g[n] * (x - mn[n]) * rsqrtf(vr[n] + EPSBN) + b[n];
  Y[m * (long)ldy + n] = (h16_t)y;
}

// ---------------- WMMA GEMM: C[M,Np] = A[M,K] * BT[Np,K]^T (+bias, optional relu)
// A: f16 row-major, row stride K (K % 32 == 0).  BT: f16 [Np][K] (weights transposed).
// 256 threads = 8 waves.  NWN waves along N, 8/NWN along M; each wave owns a
// 16(M) x (NSUB*16)(N) tile.  Grid exactly tiles M and Np -> no guards, EXEC
// stays all-ones through every WMMA.  The k-loop is ping-pong unrolled x2:
// each half consumes one fragment slot and immediately re-issues that slot's
// loads for fragment i+2 (no register copies).  Requires K/32 even or == 1.
// All addresses are u64 byte addresses loaded through AS(1) pointers.
template <int NSUB, int NWN>
__global__ void __launch_bounds__(256)
k_gemm_wmma(const h16_t* __restrict__ A, const h16_t* __restrict__ BT,
            const float* __restrict__ bias, float* __restrict__ C,
            int K, int ldc, int relu) {
  constexpr int MWAVES = 8 / NWN;
  const int lane  = threadIdx.x & 31;
  const int wid   = threadIdx.x >> 5;
  const int l15   = lane & 15;
  const int khalf = lane >> 4;
  const int mbase = blockIdx.x * (16 * MWAVES) + (wid % MWAVES) * 16;
  const int nbase = blockIdx.y * (16 * NSUB * NWN) + (wid / MWAVES) * (16 * NSUB);

  // A fragment (16-bit A 16x32 ISA layout): lane holds K = khalf*8..+7 and
  // 16+khalf*8..+7 of row (mbase + l15).  u64 byte addresses, khalf pre-added.
  unsigned long long pA =
      (unsigned long long)(A + (long)(mbase + l15) * K + khalf * 8);
  unsigned long long pB[NSUB];
#pragma unroll
  for (int t = 0; t < NSUB; ++t)
    pB[t] = (unsigned long long)(BT + (long)(nbase + t * 16 + l15) * K + khalf * 8);

  v8f acc[NSUB];
#pragma unroll
  for (int t = 0; t < NSUB; ++t)
#pragma unroll
    for (int i = 0; i < 8; ++i) acc[t][i] = 0.f;

  // AS(1) 16-byte loads: A halves at +0 and +32B (16 halves), B at +0 and +32B
  auto ldA = [&](unsigned long long p, v16h& a) {
    ((f4_t*)&a)[0] = *(gcf4p)(p);
    ((f4_t*)&a)[1] = *(gcf4p)(p + 32);
  };
  auto ldB = [&](unsigned long long p, v16h& b) {
    ((f4_t*)&b)[0] = *(gcf4p)(p);
    ((f4_t*)&b)[1] = *(gcf4p)(p + 32);
  };

  v16h aA, aB;
  v16h bA[NSUB], bB[NSUB];

  const int NI = K >> 5;  // number of 32-wide k fragments (even, or 1)

  // fragment 0 -> slot A
  ldA(pA, aA);
#pragma unroll
  for (int t = 0; t < NSUB; ++t) ldB(pB[t], bA[t]);

  if (NI == 1) {
#pragma unroll
    for (int t = 0; t < NSUB; ++t)
      acc[t] = __builtin_amdgcn_wmma_f32_16x16x32_f16(
          false, aA, false, bA[t], (short)0, acc[t], false, false);
  } else {
    // fragment 1 -> slot B (fragment stride = 32 halves = 64 bytes)
    ldA(pA + 64, aB);
#pragma unroll
    for (int t = 0; t < NSUB; ++t) ldB(pB[t] + 64, bB[t]);
    pA += 128;
#pragma unroll
    for (int t = 0; t < NSUB; ++t) pB[t] += 128;

    for (int i = 0; i < NI - 2; i += 2) {
      // consume slot A (frag i), refill it with frag i+2
#pragma unroll
      for (int t = 0; t < NSUB; ++t)
        acc[t] = __builtin_amdgcn_wmma_f32_16x16x32_f16(
            false, aA, false, bA[t], (short)0, acc[t], false, false);
      ldA(pA, aA);
#pragma unroll
      for (int t = 0; t < NSUB; ++t) ldB(pB[t], bA[t]);
      // consume slot B (frag i+1), refill it with frag i+3
#pragma unroll
      for (int t = 0; t < NSUB; ++t)
        acc[t] = __builtin_amdgcn_wmma_f32_16x16x32_f16(
            false, aB, false, bB[t], (short)0, acc[t], false, false);
      ldA(pA + 64, aB);
#pragma unroll
      for (int t = 0; t < NSUB; ++t) ldB(pB[t] + 64, bB[t]);
      pA += 128;
#pragma unroll
      for (int t = 0; t < NSUB; ++t) pB[t] += 128;
    }
    // drain the last two fragments
#pragma unroll
    for (int t = 0; t < NSUB; ++t)
      acc[t] = __builtin_amdgcn_wmma_f32_16x16x32_f16(
          false, aA, false, bA[t], (short)0, acc[t], false, false);
#pragma unroll
    for (int t = 0; t < NSUB; ++t)
      acc[t] = __builtin_amdgcn_wmma_f32_16x16x32_f16(
          false, aB, false, bB[t], (short)0, acc[t], false, false);
  }

  // C layout: VGPR r -> rows (mbase+r) for lanes 0-15, (mbase+8+r) for 16-31
  gfp  gC    = (gfp)(unsigned long long)C;
  gcfp gbias = (gcfp)(unsigned long long)bias;
  const int mrow0 = mbase + ((lane < 16) ? 0 : 8);
#pragma unroll
  for (int t = 0; t < NSUB; ++t) {
    int n = nbase + t * 16 + l15;
    float bv = bias ? gbias[n] : 0.f;
#pragma unroll
    for (int r = 0; r < 8; ++r) {
      float v = acc[t][r] + bv;
      if (relu) v = fmaxf(v, 0.f);
      gC[(long)(mrow0 + r) * ldc + n] = v;
    }
  }
}

// ---------------- LSTM gate update for one timestep
__global__ void k_lstm_gate(const float* __restrict__ XZ, int t, int T,
                            const float* __restrict__ HZ,
                            float* __restrict__ Cst, h16_t* __restrict__ Hst,
                            h16_t* __restrict__ Hseq16, float* __restrict__ Hseq32,
                            int Bn, int u) {
  ASSUME_GLOBAL(XZ); ASSUME_GLOBAL(HZ); ASSUME_GLOBAL(Cst); ASSUME_GLOBAL(Hst);
  long idx = (long)blockIdx.x * 256 + threadIdx.x;
  if (idx >= (long)Bn * u) return;
  int b = (int)(idx / u), j = (int)(idx % u);
  const float* z0 = XZ + ((long)b * T + t) * (4 * u);
  const float* hz = HZ + (long)b * (4 * u);
  float zi = z0[j] + hz[j];
  float zf = z0[u + j] + hz[u + j];
  float zg = z0[2 * u + j] + hz[2 * u + j];
  float zo = z0[3 * u + j] + hz[3 * u + j];
  float si = 1.f / (1.f + __expf(-zi));
  float sf = 1.f / (1.f + __expf(-zf));
  float so = 1.f / (1.f + __expf(-zo));
  float c = Cst[idx];
  c = sf * c + si * tanhf(zg);
  float h = so * tanhf(c);
  Cst[idx] = c;
  Hst[idx] = (h16_t)h;
  if (Hseq16) { ASSUME_GLOBAL(Hseq16); Hseq16[((long)b * T + t) * u + j] = (h16_t)h; }
  if (Hseq32) { ASSUME_GLOBAL(Hseq32); Hseq32[((long)b * T + t) * u + j] = h; }
}

// ---------------- small attention core: per (batch, head), T=30, kd=64
__global__ void k_attn(const float* __restrict__ Q, const float* __restrict__ Ks,
                       const float* __restrict__ Vs, float* __restrict__ O,
                       int T, int D) {
  ASSUME_GLOBAL(Q); ASSUME_GLOBAL(Ks); ASSUME_GLOBAL(Vs); ASSUME_GLOBAL(O);
  __shared__ float Kt[30 * 64];
  __shared__ float Vt[30 * 64];
  int bb = blockIdx.x, h = blockIdx.y;
  long base = (long)bb * T * D + h * 64;
  for (int i = threadIdx.x; i < T * 64; i += 32) {
    int s = i >> 6, d = i & 63;
    Kt[i] = Ks[base + (long)s * D + d];
    Vt[i] = Vs[base + (long)s * D + d];
  }
  __syncthreads();
  int t = threadIdx.x;
  if (t < T) {
    float q[64];
#pragma unroll
    for (int d = 0; d < 64; ++d) q[d] = Q[base + (long)t * D + d];
    float sc[30];
    float mx = -1e30f;
    for (int s = 0; s < T; ++s) {
      float a = 0.f;
#pragma unroll
      for (int d = 0; d < 64; ++d) a += q[d] * Kt[s * 64 + d];
      a *= 0.125f;  // 1/sqrt(64)
      sc[s] = a;
      if (a > mx) mx = a;
    }
    float sum = 0.f;
    for (int s = 0; s < T; ++s) { sc[s] = __expf(sc[s] - mx); sum += sc[s]; }
    float inv = 1.f / sum;
#pragma unroll
    for (int d = 0; d < 64; ++d) {
      float a = 0.f;
      for (int s = 0; s < T; ++s) a += sc[s] * Vt[s * 64 + d];
      O[base + (long)t * D + d] = a * inv;
    }
  }
}

// ---------------- layernorm with residual: y = LN(Xa + Xres) -> f32 (opt) + f16
__global__ void k_ln(const float* __restrict__ Xa, const float* __restrict__ Xres,
                     const float* __restrict__ g, const float* __restrict__ b,
                     float* __restrict__ Yf, h16_t* __restrict__ Yh, int N) {
  ASSUME_GLOBAL(Xa); ASSUME_GLOBAL(Xres); ASSUME_GLOBAL(g); ASSUME_GLOBAL(b);
  ASSUME_GLOBAL(Yh);
  long row = blockIdx.x;
  __shared__ float s1[8], s2[8];
  float ls = 0.f, lq = 0.f;
  for (int n = threadIdx.x; n < N; n += 256) {
    float v = Xa[row * N + n] + Xres[row * N + n];
    ls += v; lq += v * v;
  }
  for (int o = 16; o > 0; o >>= 1) { ls += __shfl_down(ls, o); lq += __shfl_down(lq, o); }
  int lane = threadIdx.x & 31, w = threadIdx.x >> 5;
  if (lane == 0) { s1[w] = ls; s2[w] = lq; }
  __syncthreads();
  if (w == 0) {
    ls = (lane < 8) ? s1[lane] : 0.f;
    lq = (lane < 8) ? s2[lane] : 0.f;
    for (int o = 4; o > 0; o >>= 1) { ls += __shfl_down(ls, o); lq += __shfl_down(lq, o); }
    if (lane == 0) { s1[0] = ls; s2[0] = lq; }
  }
  __syncthreads();
  float mu = s1[0] / N;
  float var = s2[0] / N - mu * mu;
  float rs = rsqrtf(var + EPSBN);
  for (int n = threadIdx.x; n < N; n += 256) {
    float v = Xa[row * N + n] + Xres[row * N + n];
    float y = g[n] * (v - mu) * rs + b[n];
    if (Yf) { ASSUME_GLOBAL(Yf); Yf[row * N + n] = y; }
    Yh[row * N + n] = (h16_t)y;
  }
}

// ---------------- row softmax over first N cols (ld = ldx), write packed
__global__ void k_softmax(const float* __restrict__ X, int ldx, int N,
                          float* __restrict__ out) {
  ASSUME_GLOBAL(X); ASSUME_GLOBAL(out);
  long row = blockIdx.x;
  __shared__ float s1[8];
  float lmax = -1e30f;
  for (int n = threadIdx.x; n < N; n += 256) lmax = fmaxf(lmax, X[row * ldx + n]);
  for (int o = 16; o > 0; o >>= 1) lmax = fmaxf(lmax, __shfl_down(lmax, o));
  int lane = threadIdx.x & 31, w = threadIdx.x >> 5;
  if (lane == 0) s1[w] = lmax;
  __syncthreads();
  if (w == 0) {
    lmax = (lane < 8) ? s1[lane] : -1e30f;
    for (int o = 4; o > 0; o >>= 1) lmax = fmaxf(lmax, __shfl_down(lmax, o));
    if (lane == 0) s1[0] = lmax;
  }
  __syncthreads();
  float mx = s1[0];
  __syncthreads();
  float lsum = 0.f;
  for (int n = threadIdx.x; n < N; n += 256) lsum += __expf(X[row * ldx + n] - mx);
  for (int o = 16; o > 0; o >>= 1) lsum += __shfl_down(lsum, o);
  if (lane == 0) s1[w] = lsum;
  __syncthreads();
  if (w == 0) {
    lsum = (lane < 8) ? s1[lane] : 0.f;
    for (int o = 4; o > 0; o >>= 1) lsum += __shfl_down(lsum, o);
    if (lane == 0) s1[0] = lsum;
  }
  __syncthreads();
  float inv = 1.f / s1[0];
  for (int n = threadIdx.x; n < N; n += 256)
    out[row * N + n] = __expf(X[row * ldx + n] - mx) * inv;
}

// ===========================================================================
extern "C" void kernel_launch(void* const* d_in, const int* in_sizes, int n_in,
                              void* d_out, int out_size, void* d_ws, size_t ws_size,
                              hipStream_t stream) {
  (void)in_sizes; (void)n_in; (void)out_size; (void)ws_size;
  const int  Bn = 512, T = 30;
  const long Mtok = (long)Bn * T;  // 15360 (multiple of 128)

  const float* x = (const float*)d_in[0];
  auto P = [&](int i) { return (const float*)d_in[i]; };

  // deterministic bump allocator over workspace
  char*  base = (char*)d_ws;
  size_t off  = 0;
  auto alloc = [&](size_t bytes) -> void* {
    void* p = base + off;
    off = (off + bytes + 255) & ~(size_t)255;
    return p;
  };
  auto rup32 = [](int v) { return (v + 31) & ~31; };

  // persistent activation buffers
  h16_t* X16    = (h16_t*)alloc((size_t)Mtok * 1408 * 2);
  h16_t* S16a   = (h16_t*)alloc((size_t)Mtok * 256 * 2);
  h16_t* S16b   = (h16_t*)alloc((size_t)Mtok * 256 * 2);
  h16_t* FUSE16 = (h16_t*)alloc((size_t)Mtok * 384 * 2);
  float* C32    = (float*)alloc((size_t)Mtok * 2048 * 4);
  float* XZ32   = (float*)alloc((size_t)Mtok * 2048 * 4);
  h16_t* H16    = (h16_t*)alloc((size_t)Mtok * 512 * 2);
  float* H32    = (float*)alloc((size_t)Mtok * 512 * 4);
  float* Q32    = (float*)alloc((size_t)Mtok * 512 * 4);
  float* K32    = (float*)alloc((size_t)Mtok * 512 * 4);
  float* V32    = (float*)alloc((size_t)Mtok * 512 * 4);
  float* O32    = (float*)alloc((size_t)Mtok * 512 * 4);
  h16_t* U16    = (h16_t*)alloc((size_t)Mtok * 1024 * 2);
  h16_t* TMP16  = (h16_t*)alloc((size_t)Mtok * 512 * 2);
  float* HZ32   = (float*)alloc((size_t)Bn * 2048 * 4);
  float* hC     = (float*)alloc((size_t)Bn * 512 * 4);
  h16_t* hH16   = (h16_t*)alloc((size_t)Bn * 512 * 2);
  float* bias250= (float*)alloc(256 * 4);

  // helpers
  auto packW = [&](int idx, int K, int N) -> h16_t* {
    int Kp = rup32(K), Np = rup32(N);
    h16_t* W = (h16_t*)alloc((size_t)Kp * Np * 2);
    long tot = (long)Kp * Np;
    k_pack_w<<<dim3((unsigned)((tot + 255) / 256)), 256, 0, stream>>>(P(idx), W, K, N, Kp, Np);
    return W;
  };
  // Np must be 32, 64, or a multiple of 128; Mr a multiple of 128.
  auto gemm = [&](const h16_t* A, const h16_t* W, const float* bias, float* C,
                  long Mr, int Kp, int Np, int ldc, int relu) {
    if ((Np & 127) == 0) {
      dim3 grid((unsigned)(Mr / 64), (unsigned)(Np / 128));
      k_gemm_wmma<4, 2><<<grid, 256, 0, stream>>>(A, W, bias, C, Kp, ldc, relu);
    } else if (Np == 64) {
      dim3 grid((unsigned)(Mr / 128), 1);
      k_gemm_wmma<4, 1><<<grid, 256, 0, stream>>>(A, W, bias, C, Kp, ldc, relu);
    } else {  // Np == 32
      dim3 grid((unsigned)(Mr / 128), 1);
      k_gemm_wmma<2, 1><<<grid, 256, 0, stream>>>(A, W, bias, C, Kp, ldc, relu);
    }
  };
  auto cvt = [&](const float* X, int ldx, int c0, int K, h16_t* Y, int ldy, int co,
                 int Wout, long Mr) {
    long tot = Mr * (long)Wout;
    k_cvt_f16<<<dim3((unsigned)((tot + 255) / 256)), 256, 0, stream>>>(X, ldx, c0, K, Y, ldy, co, Wout, Mr);
  };
  auto bn = [&](const float* X, int ldx, int gi, h16_t* Y, int ldy, long Mr, int N) {
    long tot = Mr * (long)N;
    k_bn_f16<<<dim3((unsigned)((tot + 255) / 256)), 256, 0, stream>>>(
        X, ldx, P(gi), P(gi + 1), P(gi + 2), P(gi + 3), Y, ldy, Mr, N);
  };
  auto zero32 = [&](float* p, long n) {
    k_zero32<<<dim3((unsigned)((n + 255) / 256)), 256, 0, stream>>>(p, n);
  };
  auto zero16 = [&](h16_t* p, long n) {
    k_zero16<<<dim3((unsigned)((n + 255) / 256)), 256, 0, stream>>>(p, n);
  };

  // ---------------- per-token feature branches -> FUSE16 (M x 384) ----------
  {  // face cnn3: 1404 ->32(bn) ->64(bn) ->128 ->64  => FUSE[:, 0:64]
    h16_t* w1 = packW(1, 1404, 32);
    h16_t* w2 = packW(7, 32, 64);
    h16_t* w3 = packW(13, 64, 128);
    h16_t* wd = packW(15, 128, 64);
    cvt(x, 1629, 0, 1404, X16, 1408, 0, 1408, Mtok);
    gemm(X16, w1, P(2), C32, Mtok, 1408, 32, 32, 1);
    bn(C32, 32, 3, S16a, 32, Mtok, 32);
    gemm(S16a, w2, P(8), C32, Mtok, 32, 64, 64, 1);
    bn(C32, 64, 9, S16b, 64, Mtok, 64);
    gemm(S16b, w3, P(14), C32, Mtok, 64, 128, 128, 1);
    cvt(C32, 128, 0, 128, S16a, 128, 0, 128, Mtok);
    gemm(S16a, wd, P(16), C32, Mtok, 128, 64, 64, 1);
    cvt(C32, 64, 0, 64, FUSE16, 384, 0, 64, Mtok);
  }
  {  // pose cnn2: 99 ->64(bn) ->128 ->64  => FUSE[:, 64:128]
    h16_t* w1 = packW(17, 99, 64);
    h16_t* w2 = packW(23, 64, 128);
    h16_t* wd = packW(25, 128, 64);
    cvt(x, 1629, 1404, 99, X16, 128, 0, 128, Mtok);
    gemm(X16, w1, P(18), C32, Mtok, 128, 64, 64, 1);
    bn(C32, 64, 19, S16a, 64, Mtok, 64);
    gemm(S16a, w2, P(24), C32, Mtok, 64, 128, 128, 1);
    cvt(C32, 128, 0, 128, S16b, 128, 0, 128, Mtok);
    gemm(S16b, wd, P(26), C32, Mtok, 128, 64, 64, 1);
    cvt(C32, 64, 0, 64, FUSE16, 384, 64, 64, Mtok);
  }
  auto handBranch = [&](int ib, int c0, int colOut) {
    // cnn3: 63 ->64(bn) ->128(bn) ->256 ->128
    h16_t* w1 = packW(ib + 0, 63, 64);
    h16_t* w2 = packW(ib + 6, 64, 128);
    h16_t* w3 = packW(ib + 12, 128, 256);
    h16_t* wd = packW(ib + 14, 256, 128);
    cvt(x, 1629, c0, 63, X16, 64, 0, 64, Mtok);
    gemm(X16, w1, P(ib + 1), C32, Mtok, 64, 64, 64, 1);
    bn(C32, 64, ib + 2, S16a, 64, Mtok, 64);
    gemm(S16a, w2, P(ib + 7), C32, Mtok, 64, 128, 128, 1);
    bn(C32, 128, ib + 8, S16b, 128, Mtok, 128);
    gemm(S16b, w3, P(ib + 13), C32, Mtok, 128, 256, 256, 1);
    cvt(C32, 256, 0, 256, S16a, 256, 0, 256, Mtok);
    gemm(S16a, wd, P(ib + 15), C32, Mtok, 256, 128, 128, 1);
    cvt(C32, 128, 0, 128, FUSE16, 384, colOut, 128, Mtok);
  };
  handBranch(27, 1503, 128);  // left hand  => FUSE[:,128:256]
  handBranch(43, 1566, 256);  // right hand => FUSE[:,256:384]

  // fusion: 384 -> 256 relu  => S16a (M x 256)
  {
    h16_t* wf = packW(59, 384, 256);
    gemm(FUSE16, wf, P(60), C32, Mtok, 384, 256, 256, 1);
    cvt(C32, 256, 0, 256, S16a, 256, 0, 256, Mtok);
  }

  // ---------------- LSTM helper -------------------------------------------
  auto lstm = [&](const h16_t* Xin, int Kin, int wxI, int whI, int bI, int u,
                  h16_t* Hs16, float* Hs32) {
    h16_t* wx = packW(wxI, Kin, 4 * u);
    h16_t* wh = packW(whI, u, 4 * u);
    gemm(Xin, wx, P(bI), XZ32, Mtok, Kin, 4 * u, 4 * u, 0);
    zero32(hC, (long)Bn * u);
    zero16(hH16, (long)Bn * u);
    long tot = (long)Bn * u;
    for (int t = 0; t < T; ++t) {
      gemm(hH16, wh, nullptr, HZ32, Bn, u, 4 * u, 4 * u, 0);
      k_lstm_gate<<<dim3((unsigned)((tot + 255) / 256)), 256, 0, stream>>>(
          XZ32, t, T, HZ32, hC, hH16, Hs16, Hs32, Bn, u);
    }
  };

  // ---------------- attention block helper ---------------------------------
  auto attnblk = [&](h16_t* Xi16, float* Xi32, int D, int H, int ib,
                     float* q, float* k, float* v, float* o,
                     h16_t* out16, float* out32) {
    h16_t* wq  = packW(ib + 0, D, D);
    h16_t* wk  = packW(ib + 2, D, D);
    h16_t* wv  = packW(ib + 4, D, D);
    h16_t* wo  = packW(ib + 6, D, D);
    h16_t* wf1 = packW(ib + 12, D, 2 * D);
    h16_t* wf2 = packW(ib + 14, 2 * D, D);
    gemm(Xi16, wq, P(ib + 1), q, Mtok, D, D, D, 0);
    gemm(Xi16, wk, P(ib + 3), k, Mtok, D, D, D, 0);
    gemm(Xi16, wv, P(ib + 5), v, Mtok, D, D, D, 0);
    k_attn<<<dim3(Bn, H), 32, 0, stream>>>(q, k, v, o, T, D);
    cvt(o, D, 0, D, TMP16, D, 0, D, Mtok);
    gemm(TMP16, wo, P(ib + 7), C32, Mtok, D, D, D, 0);
    // out1 = LN(x + mha) -> f32 into o (reuse), f16 into TMP16
    k_ln<<<dim3((unsigned)Mtok), 256, 0, stream>>>(C32, Xi32, P(ib + 8), P(ib + 9), o, TMP16, D);
    // ffn
    gemm(TMP16, wf1, P(ib + 13), C32, Mtok, D, 2 * D, 2 * D, 1);
    cvt(C32, 2 * D, 0, 2 * D, U16, 2 * D, 0, 2 * D, Mtok);
    gemm(U16, wf2, P(ib + 15), C32, Mtok, 2 * D, D, D, 0);
    k_ln<<<dim3((unsigned)Mtok), 256, 0, stream>>>(C32, o, P(ib + 10), P(ib + 11), out32, out16, D);
  };

  // ---------------- sequence stack ----------------------------------------
  // LSTM1: 256 -> 512, seq out -> H16/H32 (ld 512)
  lstm(S16a, 256, 61, 62, 63, 512, H16, H32);
  // attn1 (d=512, 8 heads): in H16/H32, out back into H16/H32
  attnblk(H16, H32, 512, 8, 64, Q32, K32, V32, O32, H16, H32);
  // LSTM2: 512 -> 256, seq out -> S16a (ld 256) + Q32 (f32, ld 256)
  lstm(H16, 512, 80, 81, 82, 256, S16a, Q32);
  // attn2 (d=256, 4 heads)
  {
    float* q2 = K32;
    float* k2 = K32 + (size_t)Mtok * 256;
    float* v2 = V32;
    float* o2 = V32 + (size_t)Mtok * 256;
    attnblk(S16a, Q32, 256, 4, 83, q2, k2, v2, o2, S16b, O32);
  }
  // LSTM3: 256 -> 128, final hidden only (hH16 holds h_T, 512 x 128)
  lstm(S16b, 256, 99, 100, 101, 128, nullptr, nullptr);

  // ---------------- classifier + softmax -----------------------------------
  {
    h16_t* cw1 = packW(102, 128, 256);
    h16_t* cw2 = packW(108, 256, 128);
    h16_t* cw3 = packW(114, 128, 250);  // padded to Np=256
    k_pack_vec<<<dim3(1), 256, 0, stream>>>(P(115), bias250, 250, 256, 0.f);
    gemm(hH16, cw1, P(103), C32, Bn, 128, 256, 256, 1);
    bn(C32, 256, 104, S16a, 256, Bn, 256);
    gemm(S16a, cw2, P(109), C32, Bn, 256, 128, 128, 1);
    bn(C32, 128, 110, S16b, 128, Bn, 128);
    gemm(S16b, cw3, bias250, C32, Bn, 128, 256, 256, 0);
    k_softmax<<<dim3(Bn), 256, 0, stream>>>(C32, 256, 250, (float*)d_out);
  }
}
